// nestGCN_85263690760751
// MI455X (gfx1250) — matrix operand order repaired and checked by reading, not compiled, not run on hardware
//
#include <hip/hip_runtime.h>
#include <hip/hip_bf16.h>

typedef _Float16 h16 __attribute__((ext_vector_type(16)));
typedef _Float16 h2  __attribute__((ext_vector_type(2)));
typedef float    f8  __attribute__((ext_vector_type(8)));
typedef unsigned int u32x4 __attribute__((ext_vector_type(4)));
typedef int          i32x4 __attribute__((ext_vector_type(4)));
typedef int          i32x8 __attribute__((ext_vector_type(8)));

#ifndef __has_builtin
#define __has_builtin(x) 0
#endif
#if __has_builtin(__builtin_amdgcn_tensor_load_to_lds) && __has_builtin(__builtin_amdgcn_s_wait_tensorcnt)
#define USE_TDM 1
#else
#define USE_TDM 0
#endif

#define N_NODES 131072
#define G_GRAPH 2048
#define NPG     64
#define E_IN    1048576
#define E_OUT   32768
#define DIM     128
#define KTOP    8
#define KD_MAX  1024

// ---------------- weight prep: fp32 [Kd, 128] -> half transposed [128, Kd] ----------------
__global__ void wtrans_kernel(const float* __restrict__ W, _Float16* __restrict__ WT, int Kd) {
  int i = blockIdx.x * blockDim.x + threadIdx.x;
  if (i >= Kd * DIM) return;
  int k = i / DIM, c = i - k * DIM;
  WT[(size_t)c * Kd + k] = (_Float16)W[i];
}

// ---------------- x = relu(token_emb[h]) -> half ----------------
__global__ void embed_kernel(const int* __restrict__ h, const float* __restrict__ emb,
                             _Float16* __restrict__ XH) {
  int i = blockIdx.x * blockDim.x + threadIdx.x;
  if (i >= N_NODES * DIM) return;
  int node = i >> 7, d = i & 127;
  float v = emb[h[node] * DIM + d];
  XH[i] = (_Float16)(v > 0.f ? v : 0.f);
}

__global__ void fill_kernel(float* __restrict__ p, float v, int n) {
  int i = blockIdx.x * blockDim.x + threadIdx.x;
  if (i < n) p[i] = v;
}

// ---------------- WMMA GEMM: out[M,128] = Xh[M,Kd] @ W  (W given as WT[128,Kd] half) ------
// grid.x = M/16 (one 16-row stripe per block); 8 waves cover the 8 column tiles of 128.
// The block's 16xKd A stripe is DMA'd into LDS once via the Tensor Data Mover, then all
// 8 waves build A fragments from LDS (ds_load) while B streams from global (L2-resident).
__global__ void __launch_bounds__(256)
gemm_wmma_kernel(const _Float16* __restrict__ Xh, const _Float16* __restrict__ WT,
                 float* __restrict__ outF, _Float16* __restrict__ outH,
                 const float* __restrict__ bias, int Mrows, int Kd, int relu) {
  const int lane = threadIdx.x & 31;
  const int wave = threadIdx.x >> 5;        // 0..7 -> column tile
  const int m0 = blockIdx.x * 16;
  const int n0 = wave * 16;
  const int l15 = lane & 15;
  const int hi  = lane >> 4;                // 0 or 1

#if USE_TDM
  __shared__ _Float16 sA[16 * KD_MAX];      // 32 KB of the 320 KB WGP LDS
  if (threadIdx.x < 32) {
    // D# group 0: count=1 | lds_addr | global_addr[56:0] | type=2
    unsigned long long ga = (unsigned long long)(uintptr_t)Xh +
                            (unsigned long long)m0 * (unsigned)Kd * 2ull;
    unsigned ldsoff = (unsigned)(uintptr_t)(void*)sA;   // low 32 bits = LDS byte offset
    u32x4 g0;
    g0[0] = 1u;                                         // count = 1 (valid user descriptor)
    g0[1] = ldsoff;                                     // lds_addr
    g0[2] = (unsigned)(ga & 0xFFFFFFFFull);             // global_addr[31:0]
    g0[3] = (unsigned)((ga >> 32) & 0x1FFFFFFull) | (2u << 30);  // global_addr[56:32] | type=2
    // D# group 1: data_size=2B; tensor = [Mrows, Kd], tile = [16, Kd], dim0_stride = Kd
    i32x8 g1;
    g1[0] = (1 << 16);                                  // data_size code 1 => 2 bytes
    g1[1] = (int)(((unsigned)Kd & 0xFFFFu) << 16);      // tensor_dim0[15:0]
    g1[2] = (int)((((unsigned)Kd >> 16) & 0xFFFFu) |
                  (((unsigned)Mrows & 0xFFFFu) << 16)); // tensor_dim0[31:16] | tensor_dim1[15:0]
    g1[3] = (int)((((unsigned)Mrows >> 16) & 0xFFFFu) |
                  (((unsigned)Kd & 0xFFFFu) << 16));    // tensor_dim1[31:16] | tile_dim0
    g1[4] = 16;                                         // tile_dim1 = 16 rows, tile_dim2 = 0
    g1[5] = Kd;                                         // tensor_dim0_stride[31:0]
    g1[6] = 0;                                          // stride[47:32]=0, dim1_stride=0
    g1[7] = 0;
    i32x4 z4 = {0, 0, 0, 0};                            // groups 2/3 unused (2-D tensor)
#if defined(__clang_major__) && __clang_major__ >= 23
    i32x8 z8 = {0, 0, 0, 0, 0, 0, 0, 0};
    __builtin_amdgcn_tensor_load_to_lds(g0, g1, z4, z4, z8, 0);
#else
    __builtin_amdgcn_tensor_load_to_lds(g0, g1, z4, z4, 0);
#endif
    __builtin_amdgcn_s_wait_tensorcnt(0);               // TENSORcnt == 0 -> tile in LDS
  }
  __syncthreads();
  const _Float16* Arow = sA + (size_t)l15 * Kd;         // LDS tile is [16][Kd] row-major
#else
  (void)Mrows;
  const _Float16* Arow = Xh + (size_t)(m0 + l15) * Kd;
#endif
  const _Float16* Brow = WT + (size_t)(n0 + l15) * Kd;  // B: col = lane%16 (transposed W)

  f8 c = {0.f, 0.f, 0.f, 0.f, 0.f, 0.f, 0.f, 0.f};
  for (int ks = 0; ks < Kd; ks += 32) {
    h16 a, b;
#pragma unroll
    for (int i = 0; i < 8; ++i) {
      // A 16x32 f16 layout: lanes 0-15 hold K {0..7,16..23}, lanes 16-31 K {8..15,24..31}
      int ka = ks + ((i < 4) ? 2 * i : 16 + 2 * (i - 4)) + (hi ? 8 : 0);
      h2 pa = *(const h2*)(Arow + ka);
      a[2 * i] = pa.x; a[2 * i + 1] = pa.y;
      // B 32x16 f16 layout: lanes 0-15 hold K {0..7,16..23}, lanes 16-31 K {8..15,24..31}
      int kb = ks + ((i < 4) ? 2 * i : 16 + 2 * (i - 4)) + (hi ? 8 : 0);
      h2 pb = *(const h2*)(Brow + kb);
      b[2 * i] = pb.x; b[2 * i + 1] = pb.y;
    }
    c = __builtin_amdgcn_wmma_f32_16x16x32_f16(false, a, false, b, (short)0, c, false, false);
  }

  const int col = n0 + l15;
#pragma unroll
  for (int r = 0; r < 8; ++r) {
    int row = m0 + r + (hi ? 8 : 0);        // C/D: VGPR r -> M=r (lanes<16) / M=r+8
    float v = c[r];
    if (bias) v += bias[col];
    if (relu) v = v > 0.f ? v : 0.f;
    if (outF) outF[(size_t)row * DIM + col] = v;
    if (outH) outH[(size_t)row * DIM + col] = (_Float16)v;
  }
}

// ---------------- el = f@al, er = f@ar : one wave per row ----------------
__global__ void attn_logits_kernel(const float* __restrict__ F, const float* __restrict__ al,
                                   const float* __restrict__ ar, float* __restrict__ el,
                                   float* __restrict__ er, int M) {
  int row = blockIdx.x * 8 + (threadIdx.x >> 5);
  int lane = threadIdx.x & 31;
  if (row >= M) return;
  float sl = 0.f, sr = 0.f;
  for (int d = lane; d < DIM; d += 32) {
    float f = F[(size_t)row * DIM + d];
    sl += f * al[d]; sr += f * ar[d];
  }
  for (int o = 16; o; o >>= 1) { sl += __shfl_xor(sl, o, 32); sr += __shfl_xor(sr, o, 32); }
  if (lane == 0) { el[row] = sl; er[row] = sr; }
}

// ---------------- per-edge leaky-relu score + segment max over dst ----------------
__global__ void edge_score_max_kernel(const int* __restrict__ src, const int* __restrict__ dst,
                                      const float* __restrict__ el, const float* __restrict__ er,
                                      float* __restrict__ esc, float* __restrict__ m, int E) {
  int e = blockIdx.x * blockDim.x + threadIdx.x;
  if (e >= E) return;
  int s_ = src[e], d_ = dst[e];
  float sc = el[s_] + er[d_];
  sc = sc > 0.f ? sc : 0.2f * sc;
  esc[e] = sc;
  int* ai = (int*)(m + d_);                 // float atomic-max via CAS
  int cur = *ai;
  while (__int_as_float(cur) < sc) {
    int prev = atomicCAS(ai, cur, __float_as_int(sc));
    if (prev == cur) break;
    cur = prev;
  }
}

// ---------------- a = exp(sc - m[dst]); s[dst] += a ----------------
__global__ void edge_expsum_kernel(const int* __restrict__ dst, float* __restrict__ esc,
                                   const float* __restrict__ m, float* __restrict__ ssum, int E) {
  int e = blockIdx.x * blockDim.x + threadIdx.x;
  if (e >= E) return;
  int d = dst[e];
  float a = __expf(esc[e] - m[d]);
  esc[e] = a;
  atomicAdd(ssum + d, a);
}

// ---------------- out[dst] += (a/s[dst]) * f[src] : one wave per edge ----------------
__global__ void edge_aggregate_kernel(const int* __restrict__ src, const int* __restrict__ dst,
                                      const float* __restrict__ esc, const float* __restrict__ ssum,
                                      const float* __restrict__ F, float* __restrict__ ACC, int E) {
  int e = blockIdx.x * 8 + (threadIdx.x >> 5);
  int lane = threadIdx.x & 31;
  if (e >= E) return;
  int s_ = src[e], d_ = dst[e];
  float alpha = esc[e] / ssum[d_];
  const float* fs = F + (size_t)s_ * DIM;
  float* od = ACC + (size_t)d_ * DIM;
  __builtin_prefetch(fs + lane, 0, 1);      // global_prefetch_b8
#pragma unroll
  for (int i = 0; i < 4; ++i) {
    int d = lane + 32 * i;
    atomicAdd(od + d, alpha * fs[d]);
  }
}

// ---------------- relu(ACC + b) -> half ----------------
__global__ void bias_relu_half_kernel(const float* __restrict__ ACC, const float* __restrict__ b,
                                      _Float16* __restrict__ XH, int total) {
  int i = blockIdx.x * blockDim.x + threadIdx.x;
  if (i >= total) return;
  float v = ACC[i] + b[i & 127];
  XH[i] = (_Float16)(v > 0.f ? v : 0.f);
}

// ---------------- sort pooling ----------------
__global__ void rowmax_kernel(const float* __restrict__ F, float* __restrict__ RM, int M) {
  int row = blockIdx.x * 8 + (threadIdx.x >> 5);
  int lane = threadIdx.x & 31;
  if (row >= M) return;
  float mx = -3.0e38f;
  for (int d = lane; d < DIM; d += 32) { float v = F[(size_t)row * DIM + d]; mx = v > mx ? v : mx; }
  for (int o = 16; o; o >>= 1) { float t = __shfl_xor(mx, o, 32); mx = t > mx ? t : mx; }
  if (lane == 0) RM[row] = mx;
}

__global__ void top8_kernel(const float* __restrict__ RM, int* __restrict__ idx8) {
  int g = blockIdx.x;
  if (threadIdx.x != 0) return;
  const float* r = RM + (size_t)g * NPG;
  unsigned long long taken = 0ull;
  for (int t = 0; t < KTOP; ++t) {          // stable descending select (ties -> lowest index)
    float best = -3.4e38f; int bi = 0;
    for (int i = 0; i < NPG; ++i) {
      if ((taken >> i) & 1ull) continue;
      float v = r[i];
      if (v > best) { best = v; bi = i; }
    }
    taken |= 1ull << bi;
    idx8[g * KTOP + t] = bi;
  }
}

// one block of 128 threads per (graph, rank): sort the selected row ascending into XGH
__global__ void sortrow_kernel(const float* __restrict__ F, const int* __restrict__ idx8,
                               _Float16* __restrict__ XGH) {
  int gb = blockIdx.x;
  int g = gb >> 3, r = gb & 7;
  int node = g * NPG + idx8[g * KTOP + r];
  __shared__ float sd[DIM];
  int t = threadIdx.x;
  float v = F[(size_t)node * DIM + t];
  sd[t] = v;
  __syncthreads();
  int rank = 0;
  for (int j = 0; j < DIM; ++j) {
    float vj = sd[j];
    rank += (vj < v) || (vj == v && j < t);
  }
  XGH[(size_t)g * (KTOP * DIM) + r * DIM + rank] = (_Float16)v;
}

// ---------------- classifier: out[M,2] = X @ Wc + bc ----------------
__global__ void classifier_kernel(const float* __restrict__ X, const float* __restrict__ Wc,
                                  const float* __restrict__ bc, float* __restrict__ out, int total) {
  int i = blockIdx.x * blockDim.x + threadIdx.x;
  if (i >= total) return;
  int row = i >> 1, cls = i & 1;
  float acc = bc[cls];
  for (int d = 0; d < DIM; ++d) acc += X[(size_t)row * DIM + d] * Wc[d * 2 + cls];
  out[i] = acc;
}

extern "C" void kernel_launch(void* const* d_in, const int* in_sizes, int n_in,
                              void* d_out, int out_size, void* d_ws, size_t ws_size,
                              hipStream_t stream) {
  (void)in_sizes; (void)n_in; (void)out_size; (void)ws_size;
  const int*   hh   = (const int*)  d_in[0];
  const int*   gsrc = (const int*)  d_in[1];
  const int*   gdst = (const int*)  d_in[2];
  const int*   fsrc = (const int*)  d_in[3];
  const int*   fdst = (const int*)  d_in[4];
  const float* emb  = (const float*)d_in[5];
  const float* W1 = (const float*)d_in[6];  const float* al1 = (const float*)d_in[7];
  const float* ar1= (const float*)d_in[8];  const float* b1  = (const float*)d_in[9];
  const float* W2 = (const float*)d_in[10]; const float* al2 = (const float*)d_in[11];
  const float* ar2= (const float*)d_in[12]; const float* b2  = (const float*)d_in[13];
  const float* W3 = (const float*)d_in[14]; const float* al3 = (const float*)d_in[15];
  const float* ar3= (const float*)d_in[16]; const float* b3  = (const float*)d_in[17];
  const float* Wf = (const float*)d_in[18]; const float* bf  = (const float*)d_in[19];
  const float* Wl = (const float*)d_in[20]; const float* bl  = (const float*)d_in[21];
  const float* Wl1= (const float*)d_in[22]; const float* bl1 = (const float*)d_in[23];
  const float* Wc = (const float*)d_in[24]; const float* bc  = (const float*)d_in[25];
  float* outp = (float*)d_out;

  size_t off = 0;
  auto take = [&](size_t bytes) -> void* {
    void* p = (void*)((char*)d_ws + off);
    off += (bytes + 255) & ~(size_t)255;
    return p;
  };
  _Float16* XH   = (_Float16*)take((size_t)N_NODES * DIM * 2);
  float*    F    = (float*)   take((size_t)N_NODES * DIM * 4);
  float*    ACC  = (float*)   take((size_t)N_NODES * DIM * 4);
  float*    EL   = (float*)   take((size_t)N_NODES * 4);
  float*    ER   = (float*)   take((size_t)N_NODES * 4);
  float*    MX   = (float*)   take((size_t)N_NODES * 4);
  float*    SX   = (float*)   take((size_t)N_NODES * 4);
  float*    ESC  = (float*)   take((size_t)E_IN * 4);
  float*    RM   = (float*)   take((size_t)N_NODES * 4);
  int*      IDX8 = (int*)     take((size_t)G_GRAPH * KTOP * 4);
  _Float16* XGH  = (_Float16*)take((size_t)G_GRAPH * KTOP * DIM * 2);
  _Float16* W1T  = (_Float16*)take((size_t)DIM * DIM * 2);
  _Float16* W2T  = (_Float16*)take((size_t)DIM * DIM * 2);
  _Float16* WfT  = (_Float16*)take((size_t)DIM * DIM * 2);
  _Float16* WlT  = (_Float16*)take((size_t)DIM * DIM * 2);
  _Float16* Wl1T = (_Float16*)take((size_t)DIM * DIM * 2);
  _Float16* W3T  = (_Float16*)take((size_t)KTOP * DIM * DIM * 2);
  _Float16* XHg  = (_Float16*)take((size_t)G_GRAPH * DIM * 2);
  _Float16* XHg2 = (_Float16*)take((size_t)G_GRAPH * DIM * 2);

  // weight prep
  wtrans_kernel<<<(DIM * DIM + 255) / 256, 256, 0, stream>>>(W1, W1T, DIM);
  wtrans_kernel<<<(DIM * DIM + 255) / 256, 256, 0, stream>>>(W2, W2T, DIM);
  wtrans_kernel<<<(DIM * DIM + 255) / 256, 256, 0, stream>>>(Wf, WfT, DIM);
  wtrans_kernel<<<(DIM * DIM + 255) / 256, 256, 0, stream>>>(Wl, WlT, DIM);
  wtrans_kernel<<<(DIM * DIM + 255) / 256, 256, 0, stream>>>(Wl1, Wl1T, DIM);
  wtrans_kernel<<<(KTOP * DIM * DIM + 255) / 256, 256, 0, stream>>>(W3, W3T, KTOP * DIM);

  embed_kernel<<<(N_NODES * DIM) / 256, 256, 0, stream>>>(hh, emb, XH);

  auto run_gat = [&](const _Float16* Xin, const _Float16* WT, int Kd,
                     const float* al, const float* ar, const float* b,
                     const int* esrc, const int* edst, int E, int M, _Float16* Xout) {
    gemm_wmma_kernel<<<M / 16, 256, 0, stream>>>(Xin, WT, F, (_Float16*)nullptr,
                                                 (const float*)nullptr, M, Kd, 0);
    attn_logits_kernel<<<(M + 7) / 8, 256, 0, stream>>>(F, al, ar, EL, ER, M);
    fill_kernel<<<(M + 255) / 256, 256, 0, stream>>>(MX, -3.0e38f, M);
    fill_kernel<<<(M + 255) / 256, 256, 0, stream>>>(SX, 0.f, M);
    fill_kernel<<<(M * DIM + 255) / 256, 256, 0, stream>>>(ACC, 0.f, M * DIM);
    edge_score_max_kernel<<<(E + 255) / 256, 256, 0, stream>>>(esrc, edst, EL, ER, ESC, MX, E);
    edge_expsum_kernel<<<(E + 255) / 256, 256, 0, stream>>>(edst, ESC, MX, SX, E);
    edge_aggregate_kernel<<<(E + 7) / 8, 256, 0, stream>>>(esrc, edst, ESC, SX, F, ACC, E);
    bias_relu_half_kernel<<<(M * DIM + 255) / 256, 256, 0, stream>>>(ACC, b, Xout, M * DIM);
  };

  // two inner GAT layers over (g_src, g_dst)
  run_gat(XH, W1T, DIM, al1, ar1, b1, gsrc, gdst, E_IN, N_NODES, XH);
  run_gat(XH, W2T, DIM, al2, ar2, b2, gsrc, gdst, E_IN, N_NODES, XH);

  // linear_forward: F = relu(x @ Wf + bf)
  gemm_wmma_kernel<<<N_NODES / 16, 256, 0, stream>>>(XH, WfT, F, (_Float16*)nullptr, bf,
                                                     N_NODES, DIM, 1);

  // sort pooling -> XGH [G, 1024] half
  rowmax_kernel<<<(N_NODES + 7) / 8, 256, 0, stream>>>(F, RM, N_NODES);
  top8_kernel<<<G_GRAPH, 32, 0, stream>>>(RM, IDX8);
  sortrow_kernel<<<G_GRAPH * KTOP, DIM, 0, stream>>>(F, IDX8, XGH);

  // outer GAT over (fg_src, fg_dst), Kd = 1024
  run_gat(XGH, W3T, KTOP * DIM, al3, ar3, b3, fsrc, fdst, E_OUT, G_GRAPH, XHg);

  // two linears + classifier
  gemm_wmma_kernel<<<G_GRAPH / 16, 256, 0, stream>>>(XHg, WlT, F, XHg2, bl, G_GRAPH, DIM, 1);
  gemm_wmma_kernel<<<G_GRAPH / 16, 256, 0, stream>>>(XHg2, Wl1T, F, (_Float16*)nullptr, bl1,
                                                     G_GRAPH, DIM, 1);
  classifier_kernel<<<(G_GRAPH * 2 + 255) / 256, 256, 0, stream>>>(F, Wc, bc, outp, G_GRAPH * 2);
}
